// Net_87995289961147
// MI455X (gfx1250) — matrix operand order repaired and checked by reading, not compiled, not run on hardware
//
#include <hip/hip_runtime.h>
#include <stdint.h>

typedef __attribute__((ext_vector_type(16))) _Float16 v16h;
typedef __attribute__((ext_vector_type(8)))  float    v8f;

#define B_   8
#define C_   64
#define H_   128
#define W_   128
#define HW_  (H_*W_)
#define NPIX (B_*HW_)

union V16H { v16h v; _Float16 e[16]; };
union V8F  { v8f  v; float    e[8];  };

__device__ __forceinline__ float siluf(float x){ return x / (1.f + __expf(-x)); }
__device__ __forceinline__ float geluf(float x){ return 0.5f*x*(1.f + erff(x*0.70710678118654752f)); }

// ---------------- K1: depthwise 3x3 + SiLU, NCHW -> NHWC ----------------
__global__ void k_dwconv1(const float* __restrict__ ms, const float* __restrict__ w,
                          const float* __restrict__ bias, float* __restrict__ xdw)
{
    __shared__ float lds[W_*65];                 // 128 pixels x 64 ch, bank-conflict padded
    const int y  = blockIdx.x;                   // row
    const int b  = blockIdx.y;
    const int t  = threadIdx.x;                  // 256
    const int px = t & 127;
    const int c0 = (t >> 7) * 32;
    for (int c = c0; c < c0 + 32; ++c) {
        float s = bias[c];
        const float* wp = w + c*9;
        #pragma unroll
        for (int dy = -1; dy <= 1; ++dy) {
            int yy = y + dy; if (yy < 0 || yy >= H_) continue;
            #pragma unroll
            for (int dx = -1; dx <= 1; ++dx) {
                int xx = px + dx; if (xx < 0 || xx >= W_) continue;
                s += wp[(dy+1)*3 + (dx+1)] * ms[(((size_t)b*C_ + c)*H_ + yy)*W_ + xx];
            }
        }
        lds[px*65 + c] = siluf(s);
    }
    __syncthreads();
    const size_t rowbase = ((size_t)b*HW_ + (size_t)y*W_) * C_;
    for (int kk = 0; kk < 32; ++kk) {
        int gid = kk*256 + t;                    // p*64 + c
        xdw[rowbase + gid] = lds[(gid >> 6)*65 + (gid & 63)];
    }
}

// ---------------- K2: depthwise 3x3 (NHWC) + pooled partial sums ----------------
__global__ void k_dwconv2_pool(const float* __restrict__ xdw, const float* __restrict__ w,
                               const float* __restrict__ bias, float* __restrict__ tbuf,
                               float* __restrict__ pooled)
{
    __shared__ float red[4][C_];
    const int c  = threadIdx.x & 63;
    const int ry = threadIdx.x >> 6;             // 0..3
    const int y  = blockIdx.x*4 + ry;
    const int b  = blockIdx.y;
    float wl[9];
    #pragma unroll
    for (int i = 0; i < 9; ++i) wl[i] = w[c*9 + i];
    const float bb = bias[c];
    const size_t base = (size_t)b * HW_ * C_;
    float pacc = 0.f;
    for (int x = 0; x < W_; ++x) {
        float s = bb;
        #pragma unroll
        for (int dy = -1; dy <= 1; ++dy) {
            int yy = y + dy; if (yy < 0 || yy >= H_) continue;
            #pragma unroll
            for (int dx = -1; dx <= 1; ++dx) {
                int xx = x + dx; if (xx < 0 || xx >= W_) continue;
                s += wl[(dy+1)*3 + (dx+1)] * xdw[base + ((size_t)yy*W_ + xx)*C_ + c];
            }
        }
        tbuf[base + ((size_t)y*W_ + x)*C_ + c] = s;
        pacc += s;
    }
    red[ry][c] = pacc;
    __syncthreads();
    if (ry == 0) {
        float s = red[0][c] + red[1][c] + red[2][c] + red[3][c];
        atomicAdd(&pooled[b*C_ + c], s);
    }
}

// ---------------- K3: squeeze-excite MLP ----------------
__global__ void k_chanattn(const float* __restrict__ pooled,
                           const float* __restrict__ w1, const float* __restrict__ b1,
                           const float* __restrict__ w2, const float* __restrict__ b2,
                           float* __restrict__ abuf)
{
    __shared__ float hid[B_][4];
    const int t = threadIdx.x;                   // 512
    if (t < B_*4) {
        int b = t >> 2, r = t & 3;
        float s = b1[r];
        for (int c = 0; c < C_; ++c) s += (pooled[b*C_ + c] * (1.f/HW_)) * w1[r*C_ + c];
        hid[b][r] = fmaxf(s, 0.f);
    }
    __syncthreads();
    int b = t >> 6, c = t & 63;
    float s = b2[c];
    #pragma unroll
    for (int r = 0; r < 4; ++r) s += hid[b][r] * w2[c*4 + r];
    abuf[b*C_ + c] = 1.f / (1.f + __expf(-s));
}

// ---------------- K4: fused scale+GELU+LN+WMMA-proj+tanh+grid_sample+keys ----------------
// Wave = 16 pixels. Lane pair (L, L+16) owns pixel L, channels split per the
// CDNA5 16-bit A-matrix 16x32 layout: lane-half h holds K = {8h..8h+7, 16+8h..16+8h+7}.
__global__ void k_fused_main(const float* __restrict__ xdw, float* __restrict__ tfeat,
                             const float* __restrict__ abuf,
                             const float* __restrict__ lnw, const float* __restrict__ lnb,
                             const float* __restrict__ projw,
                             unsigned long long* __restrict__ keys)
{
    __shared__ float loff[8][16][4];
    const int lane = threadIdx.x & 31;
    const int wave = threadIdx.x >> 5;
    const int half = lane >> 4;
    const int M    = lane & 15;
    const int gpix = blockIdx.x*128 + wave*16 + M;
    const int b    = gpix >> 14;                 // / HW
    const int p    = gpix & (HW_ - 1);
    const int py   = p >> 7, pxx = p & 127;

    // B matrices: proj_w (3x64) padded to 16 columns. lane-half h holds
    // K = {8h..8h+7, 16+8h..16+8h+7} for column N = lane%16 (same as A).
    const int N  = lane & 15;
    const int K0 = half * 8;
    V16H b1, b2;
    #pragma unroll
    for (int s = 0; s < 16; ++s) {
        b1.e[s] = (_Float16)((N < 3) ? projw[N*C_ +      (K0 + ((s < 8) ? s : s + 8))] : 0.f);
        b2.e[s] = (_Float16)((N < 3) ? projw[N*C_ + 32 + (K0 + ((s < 8) ? s : s + 8))] : 0.f);
    }

    // load t row, channel-scale, GELU; per-lane 32 channels (A layout order)
    const float* trow = tfeat + (size_t)gpix * C_;
    float vals[32];
    float sum = 0.f, sq = 0.f;
    #pragma unroll
    for (int k = 0; k < 32; ++k) {
        int j  = k & 15;
        int cc = ((k < 16) ? 0 : 32) + half*8 + ((j < 8) ? j : j + 8);
        float v = trow[cc] * abuf[b*C_ + cc];
        v = geluf(v);
        vals[k] = v;
        sum += v; sq += v*v;
    }
    sum += __shfl_xor(sum, 16);                  // combine lane pair (wave32)
    sq  += __shfl_xor(sq, 16);
    float mean = sum * (1.f/64.f);
    float var  = sq  * (1.f/64.f) - mean*mean;
    float rstd = rsqrtf(var + 1e-6f);

    V16H a1, a2;
    #pragma unroll
    for (int k = 0; k < 32; ++k) {
        int j  = k & 15;
        int cc = ((k < 16) ? 0 : 32) + half*8 + ((j < 8) ? j : j + 8);
        float tn = (vals[k] - mean) * rstd * lnw[cc] + lnb[cc];
        if (k < 16) a1.e[k] = (_Float16)tn; else a2.e[k-16] = (_Float16)tn;
    }

    V8F cacc;
    #pragma unroll
    for (int i = 0; i < 8; ++i) cacc.e[i] = 0.f;
    cacc.v = __builtin_amdgcn_wmma_f32_16x16x32_f16(false, a1.v, false, b1.v,
                                                    (short)0, cacc.v, false, false);
    cacc.v = __builtin_amdgcn_wmma_f32_16x16x32_f16(false, a2.v, false, b2.v,
                                                    (short)0, cacc.v, false, false);

    // C layout: VGPR j -> M = j + 8*half, N = lane%16
    #pragma unroll
    for (int j = 0; j < 8; ++j)
        if (N < 3) loff[wave][j + half*8][N] = cacc.e[j];
    __syncthreads();

    float o0 = tanhf(loff[wave][M][0]);
    float o1 = tanhf(loff[wave][M][1]);
    float o2 = tanhf(loff[wave][M][2]);

    // bilinear grid_sample of x_dw at base grid + delta_p
    float gx = -1.f + 2.f * (float)pxx * (1.f/(W_-1));
    float gy = -1.f + 2.f * (float)py  * (1.f/(H_-1));
    float ix = ((gx + o0*(1.f/W_) + 1.f) * W_ - 1.f) * 0.5f;
    float iy = ((gy + o1*(1.f/H_) + 1.f) * H_ - 1.f) * 0.5f;
    float ix0 = floorf(ix), iy0 = floorf(iy);

    float acc[32];
    #pragma unroll
    for (int i = 0; i < 32; ++i) acc[i] = 0.f;
    const int cstart = half * 32;
    #pragma unroll
    for (int corner = 0; corner < 4; ++corner) {
        float xi = ix0 + (float)(corner & 1);
        float yi = iy0 + (float)(corner >> 1);
        float wgt = (1.f - fabsf(ix - xi)) * (1.f - fabsf(iy - yi));
        bool valid = (xi >= 0.f) && (xi <= (float)(W_-1)) && (yi >= 0.f) && (yi <= (float)(H_-1));
        if (valid && wgt != 0.f) {
            int cx = (int)fminf(fmaxf(xi, 0.f), (float)(W_-1));
            int cy = (int)fminf(fmaxf(yi, 0.f), (float)(H_-1));
            const float* row = xdw + (((size_t)b*HW_ + (size_t)cy*W_ + cx)*C_) + cstart;
            #pragma unroll
            for (int cI = 0; cI < 32; ++cI) acc[cI] += wgt * row[cI];
        }
    }
    float* frow = tfeat + (size_t)gpix * C_ + cstart;   // feats overwrite t in place
    #pragma unroll
    for (int cI = 0; cI < 32; ++cI) frow[cI] = acc[cI];

    if (half == 0) {                              // one key per pixel
        float ref = (float)p * (2.f/(HW_-1)) - 1.f;
        unsigned u = __float_as_uint(ref + o2);
        u = (u & 0x80000000u) ? ~u : (u | 0x80000000u);   // order-preserving float->uint
        keys[gpix] = ((unsigned long long)u << 32) | (unsigned)p;
    }
}

// ---------------- K5/K6: per-batch bitonic argsort of u64 (key|idx) ----------------
__global__ void k_bitonic_global(unsigned long long* __restrict__ keys, int k, int j)
{
    int tid = blockIdx.x*256 + threadIdx.x;
    int ib  = tid & (HW_ - 1);
    int l   = ib ^ j;
    if (l > ib) {
        int base = tid & ~(HW_ - 1);
        unsigned long long a = keys[base + ib];
        unsigned long long c = keys[base + l];
        bool asc = ((ib & k) == 0);
        if (asc ? (a > c) : (a < c)) { keys[base + ib] = c; keys[base + l] = a; }
    }
}

__global__ void k_bitonic_local(unsigned long long* __restrict__ keys, int k, int jstart)
{
    __shared__ unsigned long long sd[2048];
    const int t = threadIdx.x;                   // 1024
    const size_t base = (size_t)blockIdx.x * 2048;
    const int sb = (int)(base & (HW_ - 1));      // within-batch segment offset
    sd[t]        = keys[base + t];
    sd[t + 1024] = keys[base + t + 1024];
    __syncthreads();
    for (int j = jstart; j > 0; j >>= 1) {
        int i = ((t & ~(j - 1)) << 1) | (t & (j - 1));
        int l = i | j;
        bool asc = (((sb + i) & k) == 0);
        unsigned long long a = sd[i], c = sd[l];
        if (asc ? (a > c) : (a < c)) { sd[i] = c; sd[l] = a; }
        __syncthreads();
    }
    keys[base + t]        = sd[t];
    keys[base + t + 1024] = sd[t + 1024];
}

// ---------------- K7: gather sorted rows -> output (b, hw, c) ----------------
__global__ void k_gather(const unsigned long long* __restrict__ keys,
                         const float* __restrict__ feats, float* __restrict__ out)
{
    size_t gid = (size_t)blockIdx.x*256 + threadIdx.x;
    size_t row = gid >> 6;
    int c  = (int)(gid & 63);
    int bb = (int)(row >> 14);
    unsigned src = (unsigned)(keys[row] & 0xffffffffu);
    out[gid] = feats[(((size_t)bb << 14) + (size_t)src)*C_ + c];
}

extern "C" void kernel_launch(void* const* d_in, const int* in_sizes, int n_in,
                              void* d_out, int out_size, void* d_ws, size_t ws_size,
                              hipStream_t stream)
{
    const float* ms    = (const float*)d_in[0];
    const float* dw_w  = (const float*)d_in[1];
    const float* dw_b  = (const float*)d_in[2];
    const float* odw_w = (const float*)d_in[3];
    const float* odw_b = (const float*)d_in[4];
    const float* ca_w1 = (const float*)d_in[5];
    const float* ca_b1 = (const float*)d_in[6];
    const float* ca_w2 = (const float*)d_in[7];
    const float* ca_b2 = (const float*)d_in[8];
    const float* ln_w  = (const float*)d_in[9];
    const float* ln_b  = (const float*)d_in[10];
    const float* projw = (const float*)d_in[11];

    char* ws = (char*)d_ws;
    const size_t NB = (size_t)NPIX * C_ * sizeof(float);          // 33.5 MB
    float* xdw   = (float*)(ws);
    float* tfeat = (float*)(ws + NB);
    unsigned long long* keys = (unsigned long long*)(ws + 2*NB);  // 1 MB
    float* pooled = (float*)(ws + 2*NB + (size_t)NPIX*sizeof(unsigned long long));
    float* abuf   = pooled + B_*C_;
    float* out    = (float*)d_out;

    k_dwconv1<<<dim3(H_, B_), 256, 0, stream>>>(ms, dw_w, dw_b, xdw);
    (void)hipMemsetAsync(pooled, 0, B_*C_*sizeof(float), stream);
    k_dwconv2_pool<<<dim3(H_/4, B_), 256, 0, stream>>>(xdw, odw_w, odw_b, tfeat, pooled);
    k_chanattn<<<1, 512, 0, stream>>>(pooled, ca_w1, ca_b1, ca_w2, ca_b2, abuf);
    k_fused_main<<<NPIX/128, 256, 0, stream>>>(xdw, tfeat, abuf, ln_w, ln_b, projw, keys);

    for (int k = 2; k <= HW_; k <<= 1) {
        int j = k >> 1;
        for (; j >= 2048; j >>= 1)
            k_bitonic_global<<<NPIX/256, 256, 0, stream>>>(keys, k, j);
        k_bitonic_local<<<NPIX/2048, 1024, 0, stream>>>(keys, k, j);   // j <= 1024
    }
    k_gather<<<(NPIX*(size_t)C_)/256, 256, 0, stream>>>(keys, tfeat, out);
}
